// QCNetQML_65481071405338
// MI455X (gfx1250) — matrix operand order repaired and checked by reading, not compiled, not run on hardware
//
#include <hip/hip_runtime.h>
#include <hip/hip_bf16.h>
#include <math.h>

// ---------------------------------------------------------------------------
// QCNet: quanv lookup -> conv1+pool -> conv2+pool -> fc1 (WMMA f16) ->
//        fc2/fc3/BN/fc4/quantum head.
// MI455X reasoning: workload is bandwidth bound (~200MB traffic, ~2.6 GFLOP).
// fc1 is the only real GEMM: mapped to v_wmma_f32_16x16x32_f16 (wave32),
// weights/activations converted to f16 (halves dominant weight stream).
// ---------------------------------------------------------------------------

typedef __attribute__((ext_vector_type(16))) _Float16 v16h;
typedef __attribute__((ext_vector_type(8)))  _Float16 v8h;
typedef __attribute__((ext_vector_type(8)))  float    v8f;

#define BQ 128
#define HIMG 224
#define HQ 223           // quanv output spatial
#define C1H 110          // conv1+pool output spatial
#define C2H 54           // conv2+pool output spatial
#define K1 46656         // 16*54*54
#define NPAD 128         // fc1 out padded 120 -> 128

// ---------------------------------------------------------------- quanv ----
__global__ __launch_bounds__(256)
void quanv_kernel(const float* __restrict__ x, const float* __restrict__ table,
                  float* __restrict__ Hq) {
  __shared__ float t[16];
  if (threadIdx.x < 16) t[threadIdx.x] = table[threadIdx.x];
  __syncthreads();
  size_t gid = (size_t)blockIdx.x * 256 + threadIdx.x;
  const size_t total = (size_t)BQ * HQ * HQ;
  if (gid >= total) return;
  int j = (int)(gid % HQ);
  size_t r = gid / HQ;
  int i = (int)(r % HQ);
  int b = (int)(r / HQ);
  const float* xb = x + (size_t)b * HIMG * HIMG;
  int o = i * HIMG + j;
  int idx = (xb[o]         > 0.5f ? 1 : 0)
          + (xb[o + 1]     > 0.5f ? 2 : 0)
          + (xb[o + HIMG]  > 0.5f ? 4 : 0)
          + (xb[o + HIMG+1]> 0.5f ? 8 : 0);
  Hq[gid] = t[idx];
}

// --------------------------------------------------- conv1 + relu + pool ---
__global__ __launch_bounds__(256)
void conv1_pool_kernel(const float* __restrict__ Hq, const float* __restrict__ w,
                       const float* __restrict__ bias, float* __restrict__ C1) {
  __shared__ float sw[6 * 25];
  __shared__ float sb[6];
  for (int i = threadIdx.x; i < 150; i += 256) sw[i] = w[i];
  if (threadIdx.x < 6) sb[threadIdx.x] = bias[threadIdx.x];
  __syncthreads();
  size_t gid = (size_t)blockIdx.x * 256 + threadIdx.x;
  const size_t total = (size_t)BQ * 6 * C1H * C1H;
  if (gid >= total) return;
  int xo = (int)(gid % C1H); size_t r = gid / C1H;
  int yo = (int)(r % C1H);   r /= C1H;
  int c  = (int)(r % 6);
  int b  = (int)(r / 6);
  const float* in = Hq + (size_t)b * (HQ * HQ);
  const float* wc = sw + c * 25;
  float best = -INFINITY;
  #pragma unroll
  for (int pi = 0; pi < 2; ++pi) {
    #pragma unroll
    for (int pj = 0; pj < 2; ++pj) {
      int i0 = yo + pi, j0 = xo + pj;          // conv-out coords 0..110
      float s = sb[c];
      #pragma unroll
      for (int di = 0; di < 5; ++di) {
        int rr = 2 * i0 - 1 + di;
        if (rr < 0 || rr >= HQ) continue;
        #pragma unroll
        for (int dj = 0; dj < 5; ++dj) {
          int cc = 2 * j0 - 1 + dj;
          if (cc < 0 || cc >= HQ) continue;
          s += in[rr * HQ + cc] * wc[di * 5 + dj];
        }
      }
      s = s > 0.f ? s : 0.f;
      best = fmaxf(best, s);
    }
  }
  C1[gid] = best;
}

// --------------------------------- conv2 + relu + pool -> f16 activations --
__global__ __launch_bounds__(256)
void conv2_pool_kernel(const float* __restrict__ C1, const float* __restrict__ w,
                       const float* __restrict__ bias, _Float16* __restrict__ X16) {
  __shared__ float sw[16 * 6 * 9];
  __shared__ float sb[16];
  for (int i = threadIdx.x; i < 864; i += 256) sw[i] = w[i];
  if (threadIdx.x < 16) sb[threadIdx.x] = bias[threadIdx.x];
  __syncthreads();
  size_t gid = (size_t)blockIdx.x * 256 + threadIdx.x;
  const size_t total = (size_t)BQ * 16 * C2H * C2H;
  if (gid >= total) return;
  int xo = (int)(gid % C2H); size_t r = gid / C2H;
  int yo = (int)(r % C2H);   r /= C2H;
  int c  = (int)(r % 16);
  int b  = (int)(r / 16);
  const float* in = C1 + (size_t)b * 6 * (C1H * C1H);
  float best = -INFINITY;
  #pragma unroll
  for (int pi = 0; pi < 2; ++pi) {
    #pragma unroll
    for (int pj = 0; pj < 2; ++pj) {
      int i0 = yo + pi, j0 = xo + pj;          // conv-out coords 0..54
      float s = sb[c];
      for (int ic = 0; ic < 6; ++ic) {
        const float* ip = in + ic * (C1H * C1H);
        const float* wp = sw + (c * 6 + ic) * 9;
        #pragma unroll
        for (int di = 0; di < 3; ++di) {
          int rr = 2 * i0 - 1 + di;
          if (rr < 0 || rr >= C1H) continue;
          #pragma unroll
          for (int dj = 0; dj < 3; ++dj) {
            int cc = 2 * j0 - 1 + dj;
            if (cc < 0 || cc >= C1H) continue;
            s += ip[rr * C1H + cc] * wp[di * 3 + dj];
          }
        }
      }
      s = s > 0.f ? s : 0.f;
      best = fmaxf(best, s);
    }
  }
  // flatten order matches reshape(B,-1) of (B,16,54,54)
  X16[(size_t)b * K1 + (size_t)c * (C2H * C2H) + yo * C2H + xo] = (_Float16)best;
}

// ------------------------------------------ fc1 weights f32 -> f16, padded -
__global__ __launch_bounds__(256)
void convert_w1_kernel(const float* __restrict__ W, _Float16* __restrict__ Wh) {
  size_t gid = (size_t)blockIdx.x * 256 + threadIdx.x;
  const size_t total = (size_t)NPAD * K1;
  if (gid >= total) return;
  size_t k = gid % K1;
  size_t n = gid / K1;
  Wh[gid] = (n < 120) ? (_Float16)W[n * K1 + k] : (_Float16)0.f;
}

// ------------------------------------------------- fc1: WMMA f16 GEMM ------
// D[128x128] = X[128xK] * W^T[Kx128] + b, ReLU.  One wave per 16x16 tile.
// Block = 8 waves (all 8 N-tiles of one M-tile); grid = 8 M-tiles.
__global__ __launch_bounds__(256)
void fc1_wmma_kernel(const _Float16* __restrict__ X, const _Float16* __restrict__ W,
                     const float* __restrict__ bias, float* __restrict__ H1) {
  const int lane = threadIdx.x & 31;
  const int wave = threadIdx.x >> 5;      // N tile 0..7
  const int m0 = blockIdx.x * 16;         // M tile
  const int n0 = wave * 16;
  const int hi = (lane >= 16) ? 1 : 0;

  // ISA 7.12.2 16-bit A layout: lanes 0-15 -> K {0..7, 16..23}; lanes 16-31 -> K {8..15, 24..31}
  const int arow = m0 + (lane & 15);
  const _Float16* aptr = X + (size_t)arow * K1 + (hi ? 8 : 0);
  // B layout (KxN): lane holds column n0+(lane&15); lanes 0-15 K {0..7, 16..23}, lanes 16-31 K {8..15, 24..31}
  const int bcol = n0 + (lane & 15);
  const _Float16* bptr = W + (size_t)bcol * K1 + (hi ? 8 : 0);

  v8f acc = {};
  #pragma unroll 2
  for (int k = 0; k < K1; k += 32) {
    __builtin_prefetch(bptr + k + 2048, 0, 1);   // global_prefetch_b8, weight stream
    v8h a0 = *reinterpret_cast<const v8h*>(aptr + k);
    v8h a1 = *reinterpret_cast<const v8h*>(aptr + k + 16);
    v8h b0 = *reinterpret_cast<const v8h*>(bptr + k);
    v8h b1 = *reinterpret_cast<const v8h*>(bptr + k + 16);
    v16h A, B;
    #pragma unroll
    for (int i = 0; i < 8; ++i) { A[i] = a0[i]; A[i + 8] = a1[i]; }
    #pragma unroll
    for (int i = 0; i < 8; ++i) { B[i] = b0[i]; B[i + 8] = b1[i]; }
    acc = __builtin_amdgcn_wmma_f32_16x16x32_f16(
        /*neg_a=*/false, A, /*neg_b=*/false, B,
        /*c_mod=*/(short)0, acc, /*reuse_a=*/false, /*reuse_b=*/false);
  }

  // C/D layout: VGPR r -> row m0 + r + 8*hi, col = lane&15 (+n0)
  const int col = bcol;
  const float bv = (col < 120) ? bias[col] : 0.f;
  #pragma unroll
  for (int r = 0; r < 8; ++r) {
    int row = m0 + r + 8 * hi;
    float v = acc[r] + bv;
    H1[(size_t)row * NPAD + col] = v > 0.f ? v : 0.f;
  }
}

// ---------------------- fc2 + fc3 + batchnorm + fc4 + quantum head ---------
__global__ __launch_bounds__(128)
void head_kernel(const float* __restrict__ H1,
                 const float* __restrict__ w2, const float* __restrict__ b2,
                 const float* __restrict__ w3, const float* __restrict__ b3,
                 const float* __restrict__ gamma, const float* __restrict__ beta,
                 const float* __restrict__ w4, const float* __restrict__ b4,
                 const float* __restrict__ hu, float* __restrict__ out) {
  __shared__ float sW2[84 * 120];
  __shared__ float sB2[84];
  __shared__ float sW3[4 * 84];
  __shared__ float sZ[128 * 4];
  const int m = threadIdx.x;   // batch row
  for (int i = m; i < 84 * 120; i += 128) sW2[i] = w2[i];
  for (int i = m; i < 84;       i += 128) sB2[i] = b2[i];
  for (int i = m; i < 4 * 84;   i += 128) sW3[i] = w3[i];
  __syncthreads();

  const float* h1 = H1 + (size_t)m * NPAD;
  float z0 = b3[0], z1 = b3[1], z2 = b3[2], z3 = b3[3];
  for (int j = 0; j < 84; ++j) {
    float s = sB2[j];
    const float* wj = sW2 + j * 120;
    for (int k = 0; k < 120; ++k) s += wj[k] * h1[k];
    s = s > 0.f ? s : 0.f;                      // relu(fc2)
    z0 += sW3[0 * 84 + j] * s;
    z1 += sW3[1 * 84 + j] * s;
    z2 += sW3[2 * 84 + j] * s;
    z3 += sW3[3 * 84 + j] * s;
  }
  sZ[m * 4 + 0] = z0; sZ[m * 4 + 1] = z1; sZ[m * 4 + 2] = z2; sZ[m * 4 + 3] = z3;
  __syncthreads();

  // batchnorm over batch dim (each thread recomputes small reduction)
  float hn[4]; float zz[4] = {z0, z1, z2, z3};
  #pragma unroll
  for (int c = 0; c < 4; ++c) {
    float mu = 0.f;
    for (int mm = 0; mm < 128; ++mm) mu += sZ[mm * 4 + c];
    mu *= (1.f / 128.f);
    float var = 0.f;
    for (int mm = 0; mm < 128; ++mm) {
      float d = sZ[mm * 4 + c] - mu;
      var += d * d;
    }
    var *= (1.f / 128.f);
    hn[c] = gamma[c] * (zz[c] - mu) * rsqrtf(var + 1e-5f) + beta[c];
  }

  float theta = b4[0] + w4[0] * hn[0] + w4[1] * hn[1] + w4[2] * hn[2] + w4[3] * hn[3];
  float cth = cosf(theta * 0.5f);
  float sth = sinf(theta * 0.5f);
  float u0r = hu[0], u0i = hu[1], u1r = hu[2], u1i = hu[3];
  float p0r = cth * u0r + sth * u1i;
  float p0i = cth * u0i - sth * u1r;
  float p1r = cth * u1r + sth * u0i;
  float p1i = cth * u1i - sth * u0r;
  float prob = (p1r * p1r + p1i * p1i) - (p0r * p0r + p0i * p0i);
  out[m * 2 + 0] = prob;
  out[m * 2 + 1] = 1.f - prob;
}

// ---------------------------------------------------------------------------
extern "C" void kernel_launch(void* const* d_in, const int* in_sizes, int n_in,
                              void* d_out, int out_size, void* d_ws, size_t ws_size,
                              hipStream_t stream) {
  const float* x        = (const float*)d_in[0];
  const float* qtable   = (const float*)d_in[1];
  const float* head_u   = (const float*)d_in[2];
  const float* conv1_w  = (const float*)d_in[3];
  const float* conv1_b  = (const float*)d_in[4];
  const float* conv2_w  = (const float*)d_in[5];
  const float* conv2_b  = (const float*)d_in[6];
  const float* fc1_w    = (const float*)d_in[7];
  const float* fc1_b    = (const float*)d_in[8];
  const float* fc2_w    = (const float*)d_in[9];
  const float* fc2_b    = (const float*)d_in[10];
  const float* fc3_w    = (const float*)d_in[11];
  const float* fc3_b    = (const float*)d_in[12];
  const float* bn_gamma = (const float*)d_in[13];
  const float* bn_beta  = (const float*)d_in[14];
  const float* fc4_w    = (const float*)d_in[15];
  const float* fc4_b    = (const float*)d_in[16];
  float* out = (float*)d_out;

  // Workspace layout (bytes). Hq region is dead after conv1, reused for
  // f16 activations + f16 weights.
  char* ws = (char*)d_ws;
  float*    Hq  = (float*)(ws + 0);                       // 25,461,248 B
  _Float16* X16 = (_Float16*)(ws + 0);                    // 11,943,936 B (reuse)
  _Float16* W16 = (_Float16*)(ws + 11943936);             // 11,943,936 B (reuse)
  float*    C1  = (float*)(ws + 25461248);                // 37,171,200 B
  float*    H1  = (float*)(ws + 62632448);                // 65,536 B

  {
    const size_t total = (size_t)BQ * HQ * HQ;            // 6,365,312
    quanv_kernel<<<(unsigned)((total + 255) / 256), 256, 0, stream>>>(x, qtable, Hq);
  }
  {
    const size_t total = (size_t)BQ * 6 * C1H * C1H;      // 9,292,800
    conv1_pool_kernel<<<(unsigned)((total + 255) / 256), 256, 0, stream>>>(Hq, conv1_w, conv1_b, C1);
  }
  {
    const size_t total = (size_t)NPAD * K1;               // 5,971,968 (after conv1: Hq dead)
    convert_w1_kernel<<<(unsigned)((total + 255) / 256), 256, 0, stream>>>(fc1_w, W16);
  }
  {
    const size_t total = (size_t)BQ * 16 * C2H * C2H;     // 5,971,968
    conv2_pool_kernel<<<(unsigned)((total + 255) / 256), 256, 0, stream>>>(C1, conv2_w, conv2_b, X16);
  }
  fc1_wmma_kernel<<<8, 256, 0, stream>>>(X16, W16, fc1_b, H1);
  head_kernel<<<1, 128, 0, stream>>>(H1, fc2_w, fc2_b, fc3_w, fc3_b,
                                     bn_gamma, bn_beta, fc4_w, fc4_b, head_u, out);
  (void)in_sizes; (void)n_in; (void)out_size; (void)ws_size;
}
